// ClusteringModule_8933531975685
// MI455X (gfx1250) — matrix operand order repaired and checked by reading, not compiled, not run on hardware
//
#include <hip/hip_runtime.h>
#include <stdint.h>

typedef __attribute__((ext_vector_type(16))) _Float16 v16h;
typedef __attribute__((ext_vector_type(8)))  float    v8f;
typedef __attribute__((ext_vector_type(2)))  _Float16 h2;
typedef __attribute__((ext_vector_type(4)))  _Float16 h4;

// C[M][N] = A[M][K] @ B[K][N] + bias[N].  A/B are f32 in memory, converted to
// f16 on the fly, f32 WMMA accumulate.  256 threads = 8 waves per block.
// Block tile BM x BN, K-strip of 32.  Wave grid WAVES_M x WAVES_N; each wave
// computes (BM/WAVES_M/16) x (BN/WAVES_N/16) WMMA tiles of 16x16.
template<int BM, int BN, int WAVES_M, int WAVES_N>
__global__ __launch_bounds__(256) void gemm_f32_wmma(
    const float* __restrict__ A, const float* __restrict__ B,
    const float* __restrict__ bias, float* __restrict__ C,
    int Kdim, int lda, int ldb, int ldc)
{
  constexpr int BK = 32;
  constexpr int NTHREADS = 256;
  constexpr int MT = BM / (WAVES_M * 16);          // M 16-tiles per wave
  constexpr int NT = BN / (WAVES_N * 16);          // N 16-tiles per wave
  constexpr int A_TASKS = (BM * BK) / (4 * NTHREADS);       // float4 loads / thread
  constexpr int B_TASKS = ((BK / 2) * (BN / 4)) / NTHREADS; // k-pair x col4 tasks

  __shared__ __align__(16) _Float16 As[BM][BK];    // [row][k]
  __shared__ __align__(16) _Float16 Bs[BN][BK];    // [col][k] (transposed stage)

  const int t    = threadIdx.x;
  const int wave = t >> 5;
  const int lane = t & 31;
  const int wm   = wave % WAVES_M;
  const int wn   = wave / WAVES_M;
  const int m0   = blockIdx.x * BM;
  const int n0   = blockIdx.y * BN;
  const int row16 = lane & 15;
  const int half  = lane >> 4;

  // ---- precompute staging pointers (hoists all 64-bit address math out of K loop)
  const float* aptr[A_TASKS];
  _Float16*    adst[A_TASKS];
  #pragma unroll
  for (int i = 0; i < A_TASKS; ++i) {
    int f  = t + i * NTHREADS;
    int r  = f >> 3;                 // BK/4 = 8 float4 per row
    int c4 = f & 7;
    aptr[i] = A + (size_t)(m0 + r) * lda + c4 * 4;
    adst[i] = &As[r][c4 * 4];
  }
  const float* bptr[B_TASKS];
  _Float16*    bdst[B_TASKS];
  #pragma unroll
  for (int i = 0; i < B_TASKS; ++i) {
    int f  = t + i * NTHREADS;
    int kp = f / (BN / 4);           // k-pair index 0..15
    int c4 = f % (BN / 4);
    bptr[i] = B + (size_t)(2 * kp) * ldb + n0 + c4 * 4;
    bdst[i] = &Bs[c4 * 4][2 * kp];
  }

  v8f acc[MT][NT] = {};

  for (int kk = 0; kk < Kdim; kk += BK) {
    // ---- issue ALL global loads for this K-strip (b128), then convert+store
    float4 ar[A_TASKS];
    float4 b0[B_TASKS], b1[B_TASKS];
    #pragma unroll
    for (int i = 0; i < A_TASKS; ++i) {
      ar[i] = *reinterpret_cast<const float4*>(aptr[i]);
      __builtin_prefetch(aptr[i] + 2 * BK, 0, 1);
      aptr[i] += BK;
    }
    #pragma unroll
    for (int i = 0; i < B_TASKS; ++i) {
      b0[i] = *reinterpret_cast<const float4*>(bptr[i]);
      b1[i] = *reinterpret_cast<const float4*>(bptr[i] + ldb);
      __builtin_prefetch(bptr[i] + 2 * (size_t)BK * ldb, 0, 1);
      bptr[i] += (size_t)BK * ldb;
    }

    __syncthreads();   // previous iteration's fragment reads complete

    #pragma unroll
    for (int i = 0; i < A_TASKS; ++i) {
      h4 v = { (_Float16)ar[i].x, (_Float16)ar[i].y,
               (_Float16)ar[i].z, (_Float16)ar[i].w };
      *reinterpret_cast<h4*>(adst[i]) = v;         // ds_store_b64
    }
    #pragma unroll
    for (int i = 0; i < B_TASKS; ++i) {
      const float* lo = reinterpret_cast<const float*>(&b0[i]);
      const float* hi = reinterpret_cast<const float*>(&b1[i]);
      #pragma unroll
      for (int j = 0; j < 4; ++j) {
        h2 v = { (_Float16)lo[j], (_Float16)hi[j] };
        *reinterpret_cast<h2*>(bdst[i] + j * BK) = v;  // ds_store_b32 (k-pair)
      }
    }

    __syncthreads();

    // ---- fragments + WMMA
    v16h afrag[MT];
    #pragma unroll
    for (int mt = 0; mt < MT; ++mt) {
      uint32_t* ap = reinterpret_cast<uint32_t*>(&afrag[mt]);
      const int arow = wm * (MT * 16) + mt * 16 + row16;
      #pragma unroll
      for (int v = 0; v < 8; ++v) {   // ISA A layout: pairs of consecutive K
        int kb = (v < 4) ? (v * 2 + half * 8) : (16 + (v - 4) * 2 + half * 8);
        ap[v] = *reinterpret_cast<const uint32_t*>(&As[arow][kb]);
      }
    }
    #pragma unroll
    for (int nt = 0; nt < NT; ++nt) {
      v16h bfrag;
      uint32_t* bp = reinterpret_cast<uint32_t*>(&bfrag);
      const int bcol = wn * (NT * 16) + nt * 16 + row16;
      #pragma unroll
      for (int v = 0; v < 8; ++v) {   // ISA B layout: pairs of consecutive K (same as A)
        bp[v] = *reinterpret_cast<const uint32_t*>(&Bs[bcol][(v < 4) ? (v * 2 + half * 8) : (16 + (v - 4) * 2 + half * 8)]);
      }
      #pragma unroll
      for (int mt = 0; mt < MT; ++mt) {
        acc[mt][nt] = __builtin_amdgcn_wmma_f32_16x16x32_f16(
            false, afrag[mt], false, bfrag, (short)0, acc[mt][nt], false, false);
      }
    }
    __syncthreads();
  }

  // ---- epilogue: C/D layout — VGPR d, lane l: row = d + 8*(l>>4), col = l&15
  #pragma unroll
  for (int nt = 0; nt < NT; ++nt) {
    int col = n0 + wn * (NT * 16) + nt * 16 + row16;
    float bv = bias[col];
    #pragma unroll
    for (int mt = 0; mt < MT; ++mt) {
      #pragma unroll
      for (int d = 0; d < 8; ++d) {
        int row = m0 + wm * (MT * 16) + mt * 16 + d + 8 * half;
        C[(size_t)row * ldc + col] = acc[mt][nt][d] + bv;
      }
    }
  }
}

#define NROWS 2048
#define KCLUST 100
#define NDF2 64

// One 128-thread block per (b,sn) row: Student-t (alpha=1 -> s_un = 1/(1+norm)),
// row-normalize over 100 clusters, first-index argmax.
__global__ __launch_bounds__(128) void cluster_kernel(
    const float* __restrict__ E, const float* __restrict__ cent,
    float* __restrict__ s_out, float* __restrict__ c_out)
{
  __shared__ float erow[NDF2];
  __shared__ float su[128];
  __shared__ float ssum[128];
  __shared__ float smax[128];
  __shared__ int   sidx[128];

  const int row = blockIdx.x;
  const int t = threadIdx.x;

  if (t < NDF2) erow[t] = E[(size_t)row * NDF2 + t];
  __syncthreads();

  float myv = 0.0f;
  if (t < KCLUST) {
    float d2 = 0.0f;
    #pragma unroll
    for (int j = 0; j < NDF2; ++j) {
      float df = erow[j] - cent[(size_t)t * NDF2 + j];
      d2 += df * df;
    }
    float nrm = sqrtf(d2);
    myv = 1.0f / (1.0f + nrm);
  }
  su[t]   = myv;
  ssum[t] = myv;
  smax[t] = (t < KCLUST) ? myv : -1.0f;
  sidx[t] = t;
  __syncthreads();

  for (int off = 64; off > 0; off >>= 1) {
    if (t < off) {
      ssum[t] += ssum[t + off];
      float a = smax[t], b = smax[t + off];
      if (b > a || (b == a && sidx[t + off] < sidx[t])) {  // first-index argmax
        smax[t] = b;
        sidx[t] = sidx[t + off];
      }
    }
    __syncthreads();
  }

  float inv = 1.0f / ssum[0];
  if (t < KCLUST) s_out[(size_t)row * KCLUST + t] = su[t] * inv;
  if (t == 0)     c_out[row] = (float)sidx[0];
}

extern "C" void kernel_launch(void* const* d_in, const int* in_sizes, int n_in,
                              void* d_out, int out_size, void* d_ws, size_t ws_size,
                              hipStream_t stream) {
  const float* z    = (const float*)d_in[0];  // [2048][12000]
  const float* W1   = (const float*)d_in[1];  // [12000][1024]
  const float* b1   = (const float*)d_in[2];  // [1024]
  const float* W2   = (const float*)d_in[3];  // [1024][64]
  const float* b2   = (const float*)d_in[4];  // [64]
  const float* cent = (const float*)d_in[5];  // [100][64]

  float* out   = (float*)d_out;
  float* e_out = out;                            // 2048*64
  float* s_out = out + (size_t)NROWS * NDF2;     // 2048*100
  float* c_out = s_out + (size_t)NROWS * KCLUST; // 2048

  float* e1 = (float*)d_ws;                      // 2048*1024 f32 = 8 MB

  // GEMM1: [2048 x 12000] @ [12000 x 1024] + b1 -> e1   (128x128 tiles)
  dim3 g1(NROWS / 128, 1024 / 128);
  gemm_f32_wmma<128, 128, 4, 2><<<g1, 256, 0, stream>>>(
      z, W1, b1, e1, 12000, 12000, 1024, 1024);

  // GEMM2: [2048 x 1024] @ [1024 x 64] + b2 -> e        (128x64 tiles)
  dim3 g2(NROWS / 128, NDF2 / 64);
  gemm_f32_wmma<128, 64, 8, 1><<<g2, 256, 0, stream>>>(
      e1, W2, b2, e_out, 1024, 1024, NDF2, NDF2);

  // Clustering: s and c
  cluster_kernel<<<NROWS, 128, 0, stream>>>(e_out, cent, s_out, c_out);
}
